// StableGM_13426067767697
// MI455X (gfx1250) — matrix operand order repaired and checked by reading, not compiled, not run on hardware
//
#include <hip/hip_runtime.h>
#include <hip/hip_bf16.h>

typedef __attribute__((ext_vector_type(16))) _Float16 v16h;
typedef __attribute__((ext_vector_type(8)))  _Float16 v8h;
typedef __attribute__((ext_vector_type(8)))  float    v8f;

#define N_NODES 4096
#define DIM     512
#define NSINK   10
// TAU == 1.0 in the reference, so S/tau == S.

// ---------------------------------------------------------------------------
// 1) L2-normalize rows of h_s and h_t, emit fp16 copies for the WMMA GEMM.
//    One 256-thread block per row (512 elems -> 2 per thread).
// ---------------------------------------------------------------------------
__global__ __launch_bounds__(256) void sgm_normalize_f16(
    const float* __restrict__ hs, const float* __restrict__ ht,
    _Float16* __restrict__ hs16, _Float16* __restrict__ ht16)
{
  __shared__ float red[256];
  const int row = blockIdx.x;           // 0..8191 (first 4096 = h_s)
  const int tid = threadIdx.x;
  const float* src;
  _Float16* dst;
  if (row < N_NODES) { src = hs + (size_t)row * DIM;             dst = hs16 + (size_t)row * DIM; }
  else               { src = ht + (size_t)(row - N_NODES) * DIM; dst = ht16 + (size_t)(row - N_NODES) * DIM; }

  float v0 = src[tid];
  float v1 = src[tid + 256];
  red[tid] = v0 * v0 + v1 * v1;
  __syncthreads();
  for (int off = 128; off > 0; off >>= 1) {
    if (tid < off) red[tid] += red[tid + off];
    __syncthreads();
  }
  const float scale = rsqrtf(red[0]);
  dst[tid]       = (_Float16)(v0 * scale);
  dst[tid + 256] = (_Float16)(v1 * scale);
}

// ---------------------------------------------------------------------------
// 2) Zero the row/col Sinkhorn potentials (r[4096] then c[4096], contiguous).
// ---------------------------------------------------------------------------
__global__ __launch_bounds__(256) void sgm_zero_rc(float* __restrict__ rc)
{
  rc[blockIdx.x * 256 + threadIdx.x] = 0.0f;
}

// ---------------------------------------------------------------------------
// 3) WMMA fp16 GEMM: S = hs16 * ht16^T  (both operands stored row-major,
//    B columns == h_t rows, so A and B fragment loads are symmetric).
//    One wave per block; each wave owns a 2x2 group of 16x16 tiles.
//    Software-pipelined (double-buffered fragments) so next-step loads are
//    in flight while current-step WMMAs execute.
//    Fragment k order is contiguous (k = 16*half + s) in both A 16x32 and B 32x16.
// ---------------------------------------------------------------------------
__device__ __forceinline__ v16h sgm_load_a_frag(const _Float16* p, int half)
{
  // lane<16: K = 0..15; lane>=16: K = 16..31 (matches B)
  v8h lo = *(const v8h*)(p + half * 16);
  v8h hi = *(const v8h*)(p + half * 16 + 8);
  return __builtin_shufflevector(lo, hi, 0,1,2,3,4,5,6,7,8,9,10,11,12,13,14,15);
}

struct sgm_frags { v16h a0, a1, b0, b1; };

__device__ __forceinline__ sgm_frags sgm_load_frags(
    const _Float16* a0, const _Float16* a1,
    const _Float16* b0, const _Float16* b1, int k, int half)
{
  sgm_frags f;
  f.a0 = sgm_load_a_frag(a0 + k, half);
  f.a1 = sgm_load_a_frag(a1 + k, half);
  // B (32x16, column-major per lane): lane<16 holds K=0..15 of column l16,
  // lane>=16 holds K=16..31 -> one contiguous 32-byte chunk.
  f.b0 = *(const v16h*)(b0 + k + half * 16);
  f.b1 = *(const v16h*)(b1 + k + half * 16);
  return f;
}

__global__ __launch_bounds__(32) void sgm_gemm_wmma(
    const _Float16* __restrict__ A,   // hs16 [4096][512]
    const _Float16* __restrict__ B,   // ht16 [4096][512]
    float* __restrict__ S)            // [4096][4096]  (lives in d_out)
{
  const int lane = threadIdx.x;
  const int half = lane >> 4;         // 0 or 1
  const int l16  = lane & 15;
  const int m0 = blockIdx.y * 32;     // two 16-row tiles
  const int n0 = blockIdx.x * 32;     // two 16-col tiles

  const _Float16* a0 = A + (size_t)(m0 + l16) * DIM;
  const _Float16* a1 = a0 + (size_t)16 * DIM;
  const _Float16* b0 = B + (size_t)(n0 + l16) * DIM;
  const _Float16* b1 = b0 + (size_t)16 * DIM;

  v8f acc00 = {}, acc01 = {}, acc10 = {}, acc11 = {};

  sgm_frags cur = sgm_load_frags(a0, a1, b0, b1, 0, half);
  #pragma unroll 5
  for (int k = 32; k < DIM; k += 32) {
    sgm_frags nxt = sgm_load_frags(a0, a1, b0, b1, k, half);  // prefetch
    acc00 = __builtin_amdgcn_wmma_f32_16x16x32_f16(false, cur.a0, false, cur.b0, (short)0, acc00, false, false);
    acc01 = __builtin_amdgcn_wmma_f32_16x16x32_f16(false, cur.a0, false, cur.b1, (short)0, acc01, false, false);
    acc10 = __builtin_amdgcn_wmma_f32_16x16x32_f16(false, cur.a1, false, cur.b0, (short)0, acc10, false, false);
    acc11 = __builtin_amdgcn_wmma_f32_16x16x32_f16(false, cur.a1, false, cur.b1, (short)0, acc11, false, false);
    cur = nxt;
  }
  // epilogue step (k = DIM-32 fragments already resident)
  acc00 = __builtin_amdgcn_wmma_f32_16x16x32_f16(false, cur.a0, false, cur.b0, (short)0, acc00, false, false);
  acc01 = __builtin_amdgcn_wmma_f32_16x16x32_f16(false, cur.a0, false, cur.b1, (short)0, acc01, false, false);
  acc10 = __builtin_amdgcn_wmma_f32_16x16x32_f16(false, cur.a1, false, cur.b0, (short)0, acc10, false, false);
  acc11 = __builtin_amdgcn_wmma_f32_16x16x32_f16(false, cur.a1, false, cur.b1, (short)0, acc11, false, false);

  // C/D layout: VGPR r holds row (half*8 + r) of the 16x16 tile, col = l16.
  const int colA = n0 + l16;
  const int colB = n0 + 16 + l16;
  #pragma unroll
  for (int rr = 0; rr < 8; ++rr) {
    const int rowA = m0 + half * 8 + rr;
    const int rowB = m0 + 16 + half * 8 + rr;
    S[(size_t)rowA * N_NODES + colA] = acc00[rr];
    S[(size_t)rowA * N_NODES + colB] = acc01[rr];
    S[(size_t)rowB * N_NODES + colA] = acc10[rr];
    S[(size_t)rowB * N_NODES + colB] = acc11[rr];
  }
}

// ---------------------------------------------------------------------------
// 4) Row pass: r_i = logsumexp_j(S_ij - c_j). One block per row, online LSE
//    per thread, (m,s) pair-merge tree in LDS. Coalesced row-major reads.
// ---------------------------------------------------------------------------
__global__ __launch_bounds__(256) void sgm_row_lse(
    const float* __restrict__ S, const float* __restrict__ c,
    float* __restrict__ r)
{
  __shared__ float sm[256];
  __shared__ float ss[256];
  const int row = blockIdx.x;
  const int tid = threadIdx.x;
  const float* srow = S + (size_t)row * N_NODES;

  float m = -__builtin_huge_valf(), s = 0.0f;
  for (int j = tid; j < N_NODES; j += 256) {
    float v  = srow[j] - c[j];
    float nm = fmaxf(m, v);
    s = s * __expf(m - nm) + __expf(v - nm);
    m = nm;
  }
  sm[tid] = m; ss[tid] = s;
  __syncthreads();
  for (int off = 128; off > 0; off >>= 1) {
    if (tid < off) {
      float m2 = sm[tid + off], s2 = ss[tid + off];
      float nm = fmaxf(sm[tid], m2);
      ss[tid] = ss[tid] * __expf(sm[tid] - nm) + s2 * __expf(m2 - nm);
      sm[tid] = nm;
    }
    __syncthreads();
  }
  if (tid == 0) r[row] = sm[0] + __logf(ss[0]);
}

// ---------------------------------------------------------------------------
// 5) Column pass, stage 1: partial LSE over 64-row groups, coalesced
//    row-major streaming (each thread owns one column within a 256-col panel).
//    grid = (16 column panels, 64 row groups).
// ---------------------------------------------------------------------------
#define ROWG      64
#define ROWS_PG   (N_NODES / ROWG)   // 64

__global__ __launch_bounds__(256) void sgm_col_lse_partial(
    const float* __restrict__ S, const float* __restrict__ r,
    float* __restrict__ pm, float* __restrict__ ps)
{
  const int col   = blockIdx.x * 256 + threadIdx.x;
  const int g     = blockIdx.y;
  const int row0  = g * ROWS_PG;

  float m = -__builtin_huge_valf(), s = 0.0f;
  for (int i = 0; i < ROWS_PG; ++i) {
    const int row = row0 + i;
    float v  = S[(size_t)row * N_NODES + col] - r[row];
    float nm = fmaxf(m, v);
    s = s * __expf(m - nm) + __expf(v - nm);
    m = nm;
  }
  pm[(size_t)g * N_NODES + col] = m;
  ps[(size_t)g * N_NODES + col] = s;
}

// 5b) Column pass, stage 2: merge the 64 partials per column.
__global__ __launch_bounds__(256) void sgm_col_lse_combine(
    const float* __restrict__ pm, const float* __restrict__ ps,
    float* __restrict__ c)
{
  const int col = blockIdx.x * 256 + threadIdx.x;
  float m = -__builtin_huge_valf(), s = 0.0f;
  for (int g = 0; g < ROWG; ++g) {
    float m2 = pm[(size_t)g * N_NODES + col];
    float s2 = ps[(size_t)g * N_NODES + col];
    float nm = fmaxf(m, m2);
    s = s * __expf(m - nm) + s2 * __expf(m2 - nm);
    m = nm;
  }
  c[col] = m + __logf(s);
}

// ---------------------------------------------------------------------------
// 6) Final: out_ij = exp(S_ij - r_i - c_j), in place on d_out.
// ---------------------------------------------------------------------------
__global__ __launch_bounds__(256) void sgm_final_exp(
    float* __restrict__ S, const float* __restrict__ r,
    const float* __restrict__ c)
{
  const size_t idx = (size_t)blockIdx.x * 256 + threadIdx.x;
  const int i = (int)(idx >> 12);       // /4096
  const int j = (int)(idx & 4095);
  S[idx] = __expf(S[idx] - r[i] - c[j]);
}

// ---------------------------------------------------------------------------
// Launch
// ---------------------------------------------------------------------------
extern "C" void kernel_launch(void* const* d_in, const int* in_sizes, int n_in,
                              void* d_out, int out_size, void* d_ws, size_t ws_size,
                              hipStream_t stream) {
  const float* h_s = (const float*)d_in[0];   // [1,4096,512] f32
  const float* h_t = (const float*)d_in[1];   // [1,4096,512] f32
  float* S = (float*)d_out;                    // [4096,4096] f32 (scratch + final)

  // Workspace layout (~10 MB total):
  char* ws = (char*)d_ws;
  _Float16* hs16 = (_Float16*)ws;                               // 4 MB
  _Float16* ht16 = (_Float16*)(ws + ((size_t)4 << 20));          // 4 MB
  float* r  = (float*)(ws + ((size_t)8 << 20));                  // 16 KB
  float* c  = r + N_NODES;                                       // 16 KB
  float* pm = (float*)(ws + ((size_t)8 << 20) + ((size_t)64 << 10)); // 1 MB
  float* ps = pm + (size_t)ROWG * N_NODES;                       // 1 MB

  // 1) normalize + fp16 convert (8192 rows total)
  sgm_normalize_f16<<<2 * N_NODES, 256, 0, stream>>>(h_s, h_t, hs16, ht16);

  // 2) zero potentials (r and c are contiguous: 8192 floats)
  sgm_zero_rc<<<32, 256, 0, stream>>>(r);

  // 3) similarity GEMM via WMMA (each block = 1 wave = 32x32 output)
  sgm_gemm_wmma<<<dim3(N_NODES / 32, N_NODES / 32), 32, 0, stream>>>(hs16, ht16, S);

  // 4) Sinkhorn iterations on potentials only; S stays immutable (L2-resident)
  for (int it = 0; it < NSINK; ++it) {
    sgm_row_lse<<<N_NODES, 256, 0, stream>>>(S, c, r);
    sgm_col_lse_partial<<<dim3(N_NODES / 256, ROWG), 256, 0, stream>>>(S, r, pm, ps);
    sgm_col_lse_combine<<<N_NODES / 256, 256, 0, stream>>>(pm, ps, c);
  }

  // 5) rank matrix, in place
  sgm_final_exp<<<(N_NODES * (size_t)N_NODES) / 256, 256, 0, stream>>>(S, r, c);
}
